// MultiHeadedAttention_60601988546739
// MI455X (gfx1250) — matrix operand order repaired and checked by reading, not compiled, not run on hardware
//
#include <hip/hip_runtime.h>

typedef __bf16 bf16_t;
typedef __attribute__((ext_vector_type(16))) __bf16 v16bf;
typedef __attribute__((ext_vector_type(8)))  __bf16 v8bf;
typedef __attribute__((ext_vector_type(8)))  float  v8f;

#define B_  4
#define S_  2048
#define H_  16
#define DK_ 64
#define DM_ 1024
#define M_  (B_ * S_)

// ---------------------------------------------------------------------------
// A-fragment load: lane L (row M=L%16) holds contiguous K in [kb,kb+16)
// with kb=(L/16)*16 -> two 16B chunks.
__device__ __forceinline__ v16bf ld_a_frag(const bf16_t* p) {
    v16bf r;
    ((v8bf*)&r)[0] = *(const v8bf*)(p);
    ((v8bf*)&r)[1] = *(const v8bf*)(p + 8);
    return r;
}

__device__ __forceinline__ v8f wmma_bf16(v16bf a, v16bf b, v8f c) {
    return __builtin_amdgcn_wmma_f32_16x16x32_bf16(
        /*neg_a=*/false, a, /*neg_b=*/false, b,
        /*c_mod=*/(short)0, c, /*reuse_a=*/false, /*reuse_b=*/false);
}

// CDNA5 async global->LDS copy (ASYNCcnt-tracked DMA, no VGPR round-trip).
// lds_off = low 32 bits of flat shared address.
__device__ __forceinline__ void async_copy_b128(unsigned lds_off, unsigned long long gaddr) {
    asm volatile("global_load_async_to_lds_b128 %0, %1, off"
                 :: "v"(lds_off), "v"(gaddr) : "memory");
}

// ---------------------------------------------------------------------------
// Pack mask[B,S,S] (int32, 0/1) into 1 bit per element: mbits[B*S][S/32].
// One wave32 -> one 32-bit word via ballot. Runs once over 67 MB -> 2 MB.
__global__ __launch_bounds__(256) void mask_pack_kernel(
    const int* __restrict__ mask, unsigned* __restrict__ mbits)
{
    const int lane = threadIdx.x & 31;
    const int wid  = blockIdx.x * 8 + (threadIdx.x >> 5);   // word index
    const int kw   = wid & (S_/32 - 1);                     // 0..63
    const int row  = wid >> 6;                              // b*S + q
    const int mk   = mask[(size_t)row * S_ + kw*32 + lane];
    const unsigned bal = (unsigned)__ballot(mk != 0);
    if (lane == 0) mbits[wid] = bal;
}

// ---------------------------------------------------------------------------
// GEMM: C[m,n] = sum_k X[m,k] * W[n,k] + bias[n]     (Y = X @ W^T + b)
// Tile 128x128x32; 256 threads = 8 waves; wave -> 32x64 region (2x4 WMMA accs).
// Register-prefetch pipeline: global loads for tile k+1 issue while WMMAs for
// tile k run. SCALEQ folds the 1/sqrt(D_K) attention scale into Q's epilogue.
// MODE 0: out bf16 [B,H,S,64] | MODE 1: out bf16 [B,H,64,S] | MODE 2: f32 [M,DM]
template<bool XF32, int MODE, bool SCALEQ>
__global__ __launch_bounds__(256) void gemm_bias_kernel(
    const float* __restrict__ Xf, const bf16_t* __restrict__ Xb,
    const float* __restrict__ W,  const float* __restrict__ bias,
    bf16_t* __restrict__ outb, float* __restrict__ outf)
{
    __shared__ alignas(64) bf16_t As[128][32];
    __shared__ alignas(64) bf16_t Bs[128][32];

    const int tid  = threadIdx.x;
    const int lane = tid & 31, wave = tid >> 5;
    const int half = lane >> 4, l16 = lane & 15;
    const int wm = wave & 3, wn = wave >> 2;          // 4x2 wave grid
    const int m0 = blockIdx.y * 128, n0 = blockIdx.x * 128;

    const int lrow = tid >> 1;                         // 0..127
    const int lcol = (tid & 1) * 16;                   // 0 or 16

    float4 ra[4];            // staged A (f32 path)
    uint4  rab[2];           // staged A (bf16 path)
    float4 rb[4];            // staged B (always f32 weights)

    auto load_tiles = [&](int k0) {
        if constexpr (XF32) {
            const float* src = Xf + (size_t)(m0 + lrow) * DM_ + k0 + lcol;
            #pragma unroll
            for (int i = 0; i < 4; ++i) ra[i] = *(const float4*)(src + i*4);
        } else {
            const uint4* src = (const uint4*)(Xb + (size_t)(m0 + lrow) * DM_ + k0 + lcol);
            rab[0] = src[0]; rab[1] = src[1];
        }
        const float* srw = W + (size_t)(n0 + lrow) * DM_ + k0 + lcol;
        #pragma unroll
        for (int i = 0; i < 4; ++i) rb[i] = *(const float4*)(srw + i*4);
    };

    auto store_tiles = [&]() {
        bf16_t* da = &As[lrow][lcol];
        if constexpr (XF32) {
            #pragma unroll
            for (int i = 0; i < 4; ++i) {
                da[i*4+0] = (bf16_t)ra[i].x; da[i*4+1] = (bf16_t)ra[i].y;
                da[i*4+2] = (bf16_t)ra[i].z; da[i*4+3] = (bf16_t)ra[i].w;
            }
        } else {
            uint4* d = (uint4*)da;
            d[0] = rab[0]; d[1] = rab[1];
        }
        bf16_t* db = &Bs[lrow][lcol];
        #pragma unroll
        for (int i = 0; i < 4; ++i) {
            db[i*4+0] = (bf16_t)rb[i].x; db[i*4+1] = (bf16_t)rb[i].y;
            db[i*4+2] = (bf16_t)rb[i].z; db[i*4+3] = (bf16_t)rb[i].w;
        }
    };

    v8f acc[2][4] = {};
    load_tiles(0);

    for (int k0 = 0; k0 < DM_; k0 += 32) {
        store_tiles();
        __syncthreads();

        if (k0 + 32 < DM_) load_tiles(k0 + 32);   // overlap with WMMAs below

        v16bf afrag[2], bfrag[4];
        #pragma unroll
        for (int tm = 0; tm < 2; ++tm)
            afrag[tm] = ld_a_frag(&As[wm*32 + tm*16 + l16][half*16]);
        #pragma unroll
        for (int tn = 0; tn < 4; ++tn)  // B 32x16: contiguous 16 K per half-wave
            bfrag[tn] = *(const v16bf*)&Bs[wn*64 + tn*16 + l16][half*16];

        #pragma unroll
        for (int tm = 0; tm < 2; ++tm)
            #pragma unroll
            for (int tn = 0; tn < 4; ++tn)
                acc[tm][tn] = wmma_bf16(afrag[tm], bfrag[tn], acc[tm][tn]);

        __syncthreads();
    }

    // ---- epilogue: C/D layout VGPR r -> M = r + 8*half, N = l16 ----
    #pragma unroll
    for (int tm = 0; tm < 2; ++tm) {
        #pragma unroll
        for (int tn = 0; tn < 4; ++tn) {
            const int gn = n0 + wn*64 + tn*16 + l16;
            const float bv = bias[gn];
            #pragma unroll
            for (int r = 0; r < 8; ++r) {
                const int gm = m0 + wm*32 + tm*16 + half*8 + r;
                float v = acc[tm][tn][r] + bv;
                if constexpr (SCALEQ) v *= 0.125f;   // 1/sqrt(D_K), folded into Q
                if constexpr (MODE == 2) {
                    outf[(size_t)gm * DM_ + gn] = v;
                } else {
                    const int bb = gm / S_;
                    const int s  = gm & (S_ - 1);
                    const int h  = gn >> 6;
                    const int d  = gn & 63;
                    const size_t idx = (MODE == 0)
                        ? ((((size_t)(bb*H_ + h)) * S_ + s) * DK_ + d)
                        : ((((size_t)(bb*H_ + h)) * DK_ + d) * (size_t)S_ + s);
                    outb[idx] = (bf16_t)v;
                }
            }
        }
    }
}

// ---------------------------------------------------------------------------
// Flash attention: one (b,h, 128 queries) per block, 8 waves, 16 q-rows/wave.
// K/V^T chunks DMA'd to double-buffered LDS with global_load_async_to_lds_b128
// (ASYNCcnt); chunk i+1 prefetch overlaps chunk i's WMMAs + online softmax.
// Softmax denominator is computed on the matrix unit: osum = P @ 1 (all-ones
// B fragment), alpha-rescaled like O -> no cross-lane row-sum shuffles needed.
__global__ __launch_bounds__(256) void attn_kernel(
    const bf16_t* __restrict__ Qp, const bf16_t* __restrict__ Kp,
    const bf16_t* __restrict__ Vt, const unsigned* __restrict__ mbits,
    bf16_t* __restrict__ Ao)
{
    __shared__ alignas(64) bf16_t Ks[2][64][64];   // [buf][key][d]
    __shared__ alignas(64) bf16_t Vs[2][64][64];   // [buf][d][key]  (from V^T)
    __shared__ alignas(64) bf16_t Ps[8][16][64];   // per-wave P staging

    const int tid  = threadIdx.x;
    const int lane = tid & 31, wave = tid >> 5;
    const int half = lane >> 4, l16 = lane & 15;
    const int bh = blockIdx.y;
    const int bi = bh / H_, h = bh % H_;
    const int qw = blockIdx.x * 128 + wave * 16;

    // Q fragments kept in registers (pre-scaled by 1/8 in the projection)
    const bf16_t* qbase = Qp + (((size_t)bh * S_) + (qw + l16)) * DK_;
    const v16bf aq0 = ld_a_frag(qbase + half*16);
    const v16bf aq1 = ld_a_frag(qbase + 32 + half*16);

    const unsigned mlo = 1u << l16;          // bit masks for this lane's column
    const unsigned mhi = 1u << (16 + l16);

    v16bf ones;                              // all-ones B fragment for P @ 1
    #pragma unroll
    for (int i = 0; i < 16; ++i) ones[i] = (bf16_t)1.0f;

    float mrun[8];
    v8f o[4] = {};
    v8f osum = {};                           // running softmax denominator
    #pragma unroll
    for (int r = 0; r < 8; ++r) mrun[r] = -3.0e38f;

    const int lr = tid >> 2;            // 0..63
    const int lc = (tid & 3) * 16;      // 0,16,32,48  (elements)

    auto issue_chunk = [&](int sk, int buf) {   // 4 async b128 per thread
        unsigned long long kg = (unsigned long long)(uintptr_t)
            (Kp + (((size_t)bh * S_) + sk + lr) * DK_ + lc);
        unsigned long long vg = (unsigned long long)(uintptr_t)
            (Vt + (((size_t)bh * DK_) + lr) * S_ + sk + lc);
        unsigned lk = (unsigned)(uintptr_t)&Ks[buf][lr][lc];
        unsigned lv = (unsigned)(uintptr_t)&Vs[buf][lr][lc];
        async_copy_b128(lk,      kg);
        async_copy_b128(lk + 16, kg + 16);
        async_copy_b128(lv,      vg);
        async_copy_b128(lv + 16, vg + 16);
    };

    issue_chunk(0, 0);

    for (int it = 0; it < S_ / 64; ++it) {
        const int sk = it * 64;
        const int cur = it & 1;

        const bool more = (sk + 64 < S_);
        if (more) issue_chunk(sk + 64, cur ^ 1);
        // FIFO completion: <=4 outstanding means this chunk's 4 have landed.
        if (more) asm volatile("s_wait_asynccnt 0x4" ::: "memory");
        else      asm volatile("s_wait_asynccnt 0x0" ::: "memory");

        // mask words for this chunk: 64 bits per row (overlaps the wait)
        unsigned w0[8], w1[8];
        #pragma unroll
        for (int r = 0; r < 8; ++r) {
            const int qrow = qw + half*8 + r;
            const unsigned long long wp = *(const unsigned long long*)
                (mbits + (size_t)(bi * S_ + qrow) * (S_/32) + (sk >> 5));
            w0[r] = (unsigned)wp;
            w1[r] = (unsigned)(wp >> 32);
        }
        __syncthreads();

        // ---- scores: 16x64 via 4 n-tiles x 2 k-steps ----
        v8f sc[4];
        #pragma unroll
        for (int nt = 0; nt < 4; ++nt) {
            v16bf bk0 = *(const v16bf*)&Ks[cur][nt*16 + l16][half*16];
            v16bf bk1 = *(const v16bf*)&Ks[cur][nt*16 + l16][32 + half*16];
            v8f c = {};
            c = wmma_bf16(aq0, bk0, c);
            c = wmma_bf16(aq1, bk1, c);
            sc[nt] = c;
        }

        // ---- mask + per-row chunk max (scale already folded into Q) ----
        float chmax[8];
        #pragma unroll
        for (int r = 0; r < 8; ++r) chmax[r] = -3.0e38f;
        #pragma unroll
        for (int nt = 0; nt < 4; ++nt) {
            const unsigned bitm = (nt & 1) ? mhi : mlo;
            #pragma unroll
            for (int r = 0; r < 8; ++r) {
                const unsigned wsel = (nt & 2) ? w1[r] : w0[r];
                float v = sc[nt][r];
                v = (wsel & bitm) ? v : -1.0e9f;
                sc[nt][r] = v;
                chmax[r] = fmaxf(chmax[r], v);
            }
        }
        #pragma unroll
        for (int off = 8; off >= 1; off >>= 1)
            #pragma unroll
            for (int r = 0; r < 8; ++r)
                chmax[r] = fmaxf(chmax[r], __shfl_xor(chmax[r], off, 32));

        // ---- online softmax: rescale running state, exponentiate ----
        float alpha[8];
        #pragma unroll
        for (int r = 0; r < 8; ++r) {
            const float mnew = fmaxf(mrun[r], chmax[r]);
            alpha[r] = __expf(mrun[r] - mnew);
            mrun[r] = mnew;
        }
        #pragma unroll
        for (int nt = 0; nt < 4; ++nt) {
            #pragma unroll
            for (int r = 0; r < 8; ++r)
                sc[nt][r] = __expf(sc[nt][r] - mrun[r]);
        }
        #pragma unroll
        for (int r = 0; r < 8; ++r) osum[r] *= alpha[r];
        #pragma unroll
        for (int nt = 0; nt < 4; ++nt)
            #pragma unroll
            for (int r = 0; r < 8; ++r)
                o[nt][r] *= alpha[r];

        // ---- P accum-layout -> A-fragment layout via per-wave LDS ----
        #pragma unroll
        for (int nt = 0; nt < 4; ++nt)
            #pragma unroll
            for (int r = 0; r < 8; ++r)
                Ps[wave][half*8 + r][nt*16 + l16] = (bf16_t)sc[nt][r];

        const bf16_t* pp = &Ps[wave][l16][0];
        const v16bf ap0 = ld_a_frag(pp + half*16);        // keys 0..31
        const v16bf ap1 = ld_a_frag(pp + 32 + half*16);   // keys 32..63

        // ---- O += P @ V ; denominator on the matrix unit: osum += P @ 1 ----
        #pragma unroll
        for (int nt = 0; nt < 4; ++nt) {
            v16bf bv0 = *(const v16bf*)&Vs[cur][nt*16 + l16][half*16];
            v16bf bv1 = *(const v16bf*)&Vs[cur][nt*16 + l16][32 + half*16];
            o[nt] = wmma_bf16(ap0, bv0, o[nt]);
            o[nt] = wmma_bf16(ap1, bv1, o[nt]);
        }
        osum = wmma_bf16(ap0, ones, osum);
        osum = wmma_bf16(ap1, ones, osum);

        __syncthreads();   // all reads of buf `cur` done before it is re-filled
    }

    // ---- normalize and write [B,S,H*64] bf16 ----
    #pragma unroll
    for (int nt = 0; nt < 4; ++nt) {
        #pragma unroll
        for (int r = 0; r < 8; ++r) {
            const int s = qw + half*8 + r;
            const int d = nt*16 + l16;
            const float v = o[nt][r] / osum[r];
            Ao[(((size_t)bi * S_) + s) * DM_ + h*DK_ + d] = (bf16_t)v;
        }
    }
}

// ---------------------------------------------------------------------------
extern "C" void kernel_launch(void* const* d_in, const int* in_sizes, int n_in,
                              void* d_out, int out_size, void* d_ws, size_t ws_size,
                              hipStream_t stream) {
    const float* query = (const float*)d_in[0];
    const float* key   = (const float*)d_in[1];
    const float* value = (const float*)d_in[2];
    const int*   mask  = (const int*)d_in[3];
    const float* Wq = (const float*)d_in[4];  const float* bq = (const float*)d_in[5];
    const float* Wk = (const float*)d_in[6];  const float* bk = (const float*)d_in[7];
    const float* Wv = (const float*)d_in[8];  const float* bv = (const float*)d_in[9];
    const float* Wo = (const float*)d_in[10]; const float* bo = (const float*)d_in[11];
    float* out = (float*)d_out;

    const size_t proj_elems = (size_t)B_ * H_ * S_ * DK_;   // 8.4M bf16 each
    bf16_t* Qp = (bf16_t*)d_ws;
    bf16_t* Kp = Qp + proj_elems;
    bf16_t* Vt = Kp + proj_elems;
    bf16_t* Ao = Vt + proj_elems;                           // B*S*DM bf16 (same count)
    unsigned* Mb = (unsigned*)(Ao + proj_elems);            // 2 MB packed mask
    (void)in_sizes; (void)n_in; (void)out_size; (void)ws_size;

    // pack mask: 524288 words, 8 words per 256-thread block
    mask_pack_kernel<<<(B_*S_*(S_/32))/8, 256, 0, stream>>>(mask, Mb);

    dim3 gg(DM_ / 128, M_ / 128);   // (8, 64)
    gemm_bias_kernel<true, 0, true ><<<gg, 256, 0, stream>>>(query, nullptr, Wq, bq, Qp, nullptr);
    gemm_bias_kernel<true, 0, false><<<gg, 256, 0, stream>>>(key,   nullptr, Wk, bk, Kp, nullptr);
    gemm_bias_kernel<true, 1, false><<<gg, 256, 0, stream>>>(value, nullptr, Wv, bv, Vt, nullptr);

    dim3 ga(S_ / 128, B_ * H_);     // (16, 64)
    attn_kernel<<<ga, 256, 0, stream>>>(Qp, Kp, Vt, Mb, Ao);

    gemm_bias_kernel<false, 2, false><<<gg, 256, 0, stream>>>(nullptr, Ao, Wo, bo, nullptr, out);
}
